// node_o_42863773614499
// MI455X (gfx1250) — matrix operand order repaired and checked by reading, not compiled, not run on hardware
//
#include <hip/hip_runtime.h>
#include <math.h>

typedef float    v8f  __attribute__((ext_vector_type(8)));
typedef _Float16 v16h __attribute__((ext_vector_type(16)));
typedef _Float16 v8h  __attribute__((ext_vector_type(8)));

#define TT   64
#define NN   4096
#define RPB  128      // rows per block
#define WPB  8        // waves per block (wave32)
#define H1   128
#define H2   64
#define S2   136      // padded half-stride for K=128 buffers (16B aligned, bank-skewed)
#define S3   72       // padded half-stride for K=64 buffers

struct MlpP {
    const float *W1, *b1, *p1, *W2, *b2, *p2, *W3, *b3, *p3;
};
struct KParams {
    const float* a;
    MlpP  mlp[10];
    float* out;    // (64,4096)
    float* out64;  // (64,4096,64)
};

__global__ __launch_bounds__(256) void fused_traj_mlp(KParams P) {
    __shared__ _Float16 sW2 [H2 * S2];        // 64x128 f16 weights, padded
    __shared__ _Float16 sW3h[H2 * S3];        // 64x64 f16 weights (ps64 only)
    __shared__ float    sW1 [H1 * 4];         // layer1 weights, K padded to 4
    __shared__ float    sB1 [H1];
    __shared__ float    sB2 [H2];
    __shared__ float    sW3v[H2];             // layer3 weight row (out==1 MLPs)
    __shared__ float    sB3v[H2];             // layer3 bias (ps64)
    __shared__ _Float16 sAct[WPB][16 * S2];   // per-wave activations (f16)
    __shared__ float    sXin [RPB][4];        // MLP inputs, K padded to 4
    __shared__ float    sFeat[RPB][4];        // x1, y1, v, ang
    __shared__ float    sMask[RPB];
    __shared__ float    sO   [RPB][8];        // selector MLP outputs
    __shared__ float    sScal[4];             // x2, y2, v2, ang2 (block-uniform)

    const int tid   = threadIdx.x;
    const int wave  = tid >> 5;
    const int lane  = tid & 31;
    const int lm    = lane & 15;   // low-half lane id  (N / row selector)
    const int lh    = lane >> 4;   // half-of-wave selector
    const int rowB  = blockIdx.x * RPB;
    const int t     = rowB / NN;           // block-uniform (128 | 4096)
    const int nBase = rowB % NN;

    // ---------------- phase 0: per-row features ----------------
    if (tid < RPB) {
        const int n = nBase + tid;
        const float cx = P.a[((t + 1) * NN + n) * 2 + 0];
        const float cy = P.a[((t + 1) * NN + n) * 2 + 1];
        const float px = P.a[(t * NN + n) * 2 + 0];
        const float py = P.a[(t * NN + n) * 2 + 1];
        const float dx = cx - px, dy = cy - py;
        const float v  = 2.0f * sqrtf(dx * dx + dy * dy);
        const float ang = (dx == 0.0f) ? 1.57079632679489662f : atanf(dy / dx);
        const float lx = P.a[((t + 1) * NN + (NN - 1)) * 2 + 0];
        const float ly = P.a[((t + 1) * NN + (NN - 1)) * 2 + 1];
        const float ex = cx - lx, ey = cy - ly;
        sFeat[tid][0] = cx; sFeat[tid][1] = cy;
        sFeat[tid][2] = v;  sFeat[tid][3] = ang;
        sMask[tid] = (sqrtf(ex * ex + ey * ey) <= 12.0f) ? 1.0f : 0.0f;
    }
    if (tid == 0) {
        const float cx = P.a[((t + 1) * NN + (NN - 1)) * 2 + 0];
        const float cy = P.a[((t + 1) * NN + (NN - 1)) * 2 + 1];
        const float px = P.a[(t * NN + (NN - 1)) * 2 + 0];
        const float py = P.a[(t * NN + (NN - 1)) * 2 + 1];
        const float dx = cx - px, dy = cy - py;
        sScal[0] = cx; sScal[1] = cy;
        sScal[2] = 2.0f * sqrtf(dx * dx + dy * dy);
        sScal[3] = (dx == 0.0f) ? 1.57079632679489662f : atanf(dy / dx);
    }

    const int insz[10] = {1, 2, 1, 2, 1, 2, 1, 2, 4, 4};

#pragma unroll
    for (int i = 0; i < 10; ++i) {
        __syncthreads();  // everyone done with previous iteration's LDS

        // ---------------- stage weights (f32 -> f16) ----------------
        for (int idx = tid; idx < H2 * H1; idx += 256)
            sW2[(idx >> 7) * S2 + (idx & 127)] = (_Float16)P.mlp[i].W2[idx];
        if (i == 9) {
            for (int idx = tid; idx < H2 * H2; idx += 256)
                sW3h[(idx >> 6) * S3 + (idx & 63)] = (_Float16)P.mlp[i].W3[idx];
            if (tid < H2) sB3v[tid] = P.mlp[i].b3[tid];
        } else {
            if (tid < H2) sW3v[tid] = P.mlp[i].W3[tid];
        }
        for (int idx = tid; idx < H1 * 4; idx += 256) {
            const int j = idx >> 2, k = idx & 3;
            sW1[idx] = (k < insz[i]) ? P.mlp[i].W1[j * insz[i] + k] : 0.0f;
        }
        if (tid < H1) sB1[tid] = P.mlp[i].b1[tid];
        if (tid < H2) sB2[tid] = P.mlp[i].b2[tid];

        // ---------------- build MLP input vectors ----------------
        if (tid < RPB) {
            if (i < 8) {
                const int fi = i >> 1;
                sXin[tid][0] = sFeat[tid][fi];
                sXin[tid][1] = (i & 1) ? sScal[fi] : 0.0f;
                sXin[tid][2] = 0.0f; sXin[tid][3] = 0.0f;
            } else if (i == 8) {
                const float msk = sMask[tid];
                for (int f = 0; f < 4; ++f)
                    sXin[tid][f] = (msk != 0.0f) ? sO[tid][2 * f + 1]
                                                 : sO[tid][2 * f];
            } // i==9 reuses the selected features from i==8
        }
        __syncthreads();

        // ---------------- layer 1 (VALU, K<=4): 16 rows x 128 ----------------
        {
            const float p1 = P.mlp[i].p1[0];
            const int row = wave * 16 + lm;
            const float x0 = sXin[row][0], x1 = sXin[row][1];
            const float x2 = sXin[row][2], x3 = sXin[row][3];
            _Float16* actRow = &sAct[wave][lm * S2 + lh * 64];
            for (int jj = 0; jj < 8; ++jj) {
                v8h hv;
                for (int q = 0; q < 8; ++q) {
                    const int j = lh * 64 + jj * 8 + q;
                    float h = sB1[j] + x0 * sW1[j * 4 + 0] + x1 * sW1[j * 4 + 1]
                                     + x2 * sW1[j * 4 + 2] + x3 * sW1[j * 4 + 3];
                    h = (h >= 0.0f) ? h : p1 * h;
                    hv[q] = (_Float16)h;
                }
                *(v8h*)(actRow + jj * 8) = hv;   // 16B ds_store
            }
        }

        // ---------------- layer 2 (WMMA): [16x128] @ [128x64] ----------------
        v8f acc[4];
        {
            for (int j = 0; j < 4; ++j) {
                const float b = sB2[j * 16 + lm];
                for (int r = 0; r < 8; ++r) acc[j][r] = b;
            }
            for (int s = 0; s < 4; ++s) {
                const int k0 = s * 32;
                v16h afrag;
                const _Float16* ap = &sAct[wave][lm * S2 + k0 + lh * 8];
                ((v8h*)&afrag)[0] = *(const v8h*)ap;          // K = k0+lh*8 .. +7
                ((v8h*)&afrag)[1] = *(const v8h*)(ap + 16);   // K += 16
                for (int j = 0; j < 4; ++j) {
                    v16h bfrag;
                    const _Float16* bp = &sW2[(j * 16 + lm) * S2 + k0 + lh * 8];
                    ((v8h*)&bfrag)[0] = *(const v8h*)bp;        // K = k0+lh*8 .. +7
                    ((v8h*)&bfrag)[1] = *(const v8h*)(bp + 16); // K += 16
                    acc[j] = __builtin_amdgcn_wmma_f32_16x16x32_f16(
                        false, afrag, false, bfrag, (short)0, acc[j], false, false);
                }
            }
            const float p2 = P.mlp[i].p2[0];
            for (int j = 0; j < 4; ++j)
                for (int r = 0; r < 8; ++r) {
                    const float h = acc[j][r];
                    acc[j][r] = (h >= 0.0f) ? h : p2 * h;
                }
        }

        // ---------------- layer 3 ----------------
        if (i < 9) {
            // out=1: in-register dot + butterfly reduce over the 16-lane half
            const float p3 = P.mlp[i].p3[0];
            const float b3 = P.mlp[i].b3[0];
            for (int r = 0; r < 8; ++r) {
                float p = acc[0][r] * sW3v[lm]      + acc[1][r] * sW3v[16 + lm]
                        + acc[2][r] * sW3v[32 + lm] + acc[3][r] * sW3v[48 + lm];
                p += __shfl_xor(p, 1); p += __shfl_xor(p, 2);
                p += __shfl_xor(p, 4); p += __shfl_xor(p, 8);
                float o = b3 + p;
                o = (o >= 0.0f) ? o : p3 * o;
                if (lm == 0) {                       // lanes 0 and 16
                    const int m = lh * 8 + r;
                    if (i < 8) {
                        sO[wave * 16 + m][i] = o;
                    } else {                          // ps: write `out`
                        const int n = nBase + wave * 16 + m;
                        P.out[(t + 1) * NN + n] = o;
                        if (t == 0) P.out[n] = o;
                    }
                }
            }
        } else {
            // ps64: spill h2 to LDS (f16) and run a second WMMA GEMM [16x64]@[64x64]
            for (int j = 0; j < 4; ++j)
                for (int r = 0; r < 8; ++r)
                    sAct[wave][(lh * 8 + r) * S3 + j * 16 + lm] = (_Float16)acc[j][r];
            // same-wave DS ops are in-order: stores above complete before loads below
            v8f acc3[4];
            for (int j = 0; j < 4; ++j) {
                const float b = sB3v[j * 16 + lm];
                for (int r = 0; r < 8; ++r) acc3[j][r] = b;
            }
            for (int s = 0; s < 2; ++s) {
                const int k0 = s * 32;
                v16h afrag;
                const _Float16* ap = &sAct[wave][lm * S3 + k0 + lh * 8];
                ((v8h*)&afrag)[0] = *(const v8h*)ap;
                ((v8h*)&afrag)[1] = *(const v8h*)(ap + 16);
                for (int j = 0; j < 4; ++j) {
                    v16h bfrag;
                    const _Float16* bp = &sW3h[(j * 16 + lm) * S3 + k0 + lh * 8];
                    ((v8h*)&bfrag)[0] = *(const v8h*)bp;
                    ((v8h*)&bfrag)[1] = *(const v8h*)(bp + 16);
                    acc3[j] = __builtin_amdgcn_wmma_f32_16x16x32_f16(
                        false, afrag, false, bfrag, (short)0, acc3[j], false, false);
                }
            }
            const float p3 = P.mlp[9].p3[0];
            for (int j = 0; j < 4; ++j)
                for (int r = 0; r < 8; ++r) {
                    float o = acc3[j][r];
                    o = (o >= 0.0f) ? o : p3 * o;
                    const int m = lh * 8 + r;
                    const int c = j * 16 + lm;
                    const long n = nBase + wave * 16 + m;
                    P.out64[(long)(t + 1) * NN * 64 + n * 64 + c] = o;
                    if (t == 0) P.out64[n * 64 + c] = o;
                }
        }
    }
}

extern "C" void kernel_launch(void* const* d_in, const int* in_sizes, int n_in,
                              void* d_out, int out_size, void* d_ws, size_t ws_size,
                              hipStream_t stream) {
    (void)in_sizes; (void)n_in; (void)d_ws; (void)ws_size; (void)out_size;
    KParams P;
    P.a = (const float*)d_in[0];
    for (int i = 0; i < 10; ++i) {
        const int b = 1 + 9 * i;
        P.mlp[i].W1 = (const float*)d_in[b + 0];
        P.mlp[i].b1 = (const float*)d_in[b + 1];
        P.mlp[i].p1 = (const float*)d_in[b + 2];
        P.mlp[i].W2 = (const float*)d_in[b + 3];
        P.mlp[i].b2 = (const float*)d_in[b + 4];
        P.mlp[i].p2 = (const float*)d_in[b + 5];
        P.mlp[i].W3 = (const float*)d_in[b + 6];
        P.mlp[i].b3 = (const float*)d_in[b + 7];
        P.mlp[i].p3 = (const float*)d_in[b + 8];
    }
    P.out   = (float*)d_out;
    P.out64 = (float*)d_out + (long)TT * NN;   // out64 follows out in the flat buffer

    const int nRows  = (TT - 1) * NN;          // 258048
    const int blocks = nRows / RPB;            // 2016
    fused_traj_mlp<<<blocks, 256, 0, stream>>>(P);
}
